// CacheAugmentation_12979391168573
// MI455X (gfx1250) — matrix operand order repaired and checked by reading, not compiled, not run on hardware
//
#include <hip/hip_runtime.h>
#include <hip/hip_bf16.h>
#include <cstdint>

// ---------------- problem constants ----------------
#define BB     2
#define SS     1024
#define HIDD   1024
#define NHH    16
#define HDD    64
#define HOTN   1024
#define COLDN  3072
#define COMPN  512

typedef __attribute__((ext_vector_type(16))) __bf16        v16bf;
typedef __attribute__((ext_vector_type(8)))  float         v8f;
typedef __attribute__((ext_vector_type(4)))  unsigned int  u32x4;

union FragU {
    v16bf v;
    u32x4 q[2];
    unsigned short u[16];
};

__device__ __forceinline__ unsigned short f2bf(float f) {
    union { float f; unsigned int u; } x; x.f = f;
    unsigned int r = x.u + 0x7FFFu + ((x.u >> 16) & 1u);   // round-to-nearest-even
    return (unsigned short)(r >> 16);
}

// ---------------- elementwise converts ----------------
__global__ void k_f32_to_bf16(const float* __restrict__ in,
                              unsigned short* __restrict__ out, int n) {
    int i = blockIdx.x * blockDim.x + threadIdx.x;
    if (i < n) out[i] = f2bf(in[i]);
}

// in: f32 [R,C] row-major  ->  out: bf16 [C,R] row-major (transposed)
__global__ void k_transpose_bf16(const float* __restrict__ in,
                                 unsigned short* __restrict__ out, int R, int C) {
    int i = blockIdx.x * blockDim.x + threadIdx.x;
    if (i < R * C) {
        int r = i / C, c = i - r * C;
        out[c * R + r] = f2bf(in[i]);
    }
}

__global__ void k_bias_combine(const float* __restrict__ age,
                               const float* __restrict__ access,
                               float* __restrict__ out, int n) {
    int i = blockIdx.x * blockDim.x + threadIdx.x;
    if (i < n) out[i] = -0.1f * age[i] + 0.05f * access[i];
}

// ---------------- WMMA fragment loaders ----------------
// A fragment (16x32 bf16, MxK): lane m=lane&15 holds row m (hi = lane>>4).
//   elems 0..7  -> K = kk + 8*hi + i
//   elems 8..15 -> K = kk + 8*hi + 16 + (i-8)
__device__ __forceinline__ v16bf load_afrag(const unsigned short* __restrict__ base,
                                            int ldA, int row, int kk, int lane) {
    int hi = lane >> 4;
    const unsigned short* p = base + (size_t)row * ldA + kk + hi * 8;
    FragU u;
    u.q[0] = *(const u32x4*)(p);
    u.q[1] = *(const u32x4*)(p + 16);
    return u.v;
}

// B fragment (32x16 bf16, KxN) from BT [N,K] row-major: lane n=lane&15 holds col n.
//   elems 0..7 -> K = kk + 8*hi + i ; elems 8..15 -> K = kk + 8*hi + 16 + (i-8)
__device__ __forceinline__ v16bf load_bfrag(const unsigned short* __restrict__ BT,
                                            int ldB, int ncol, int kk, int lane) {
    int hi = lane >> 4;
    const unsigned short* p = BT + (size_t)ncol * ldB + kk + hi * 8;
    FragU u;
    u.q[0] = *(const u32x4*)(p);
    u.q[1] = *(const u32x4*)(p + 16);
    return u.v;
}

// ---------------- LDS-tiled, ping-pong pipelined bf16 WMMA GEMM ----------------
// Cout[M,N] (f32) = A[M,K](bf16) @ BT[N,K]^T (bf16) + bscale * bias[N]
// block = 256 threads (8 waves, 2x4 M/N wave grid); block tile 64x256.
// Per wave: 32x64 output (2x4 accum tiles). A tile (64x32) staged in LDS.
// K loop unrolled x2 with explicit ping-pong register sets (bA/bB) and
// double-buffered LDS, so no per-iteration fragment register copies.
// grid (N/256, M/64); requires M%64==0, N%256==0, K%64==0.
__global__ __launch_bounds__(256)
void k_gemm_bf16(const unsigned short* __restrict__ A,
                 const unsigned short* __restrict__ BT,
                 const float* __restrict__ bias, float bscale,
                 float* __restrict__ Cout, int M, int N, int K) {
    __shared__ unsigned short ldsA[2][64 * 32];   // 2 x 4KB

    int tid  = threadIdx.x;
    int wave = tid >> 5, lane = tid & 31;
    int wm = wave >> 2, wn = wave & 3;            // 2x4 wave grid
    int hi = lane >> 4, cl = lane & 15;
    int m0 = blockIdx.y * 64;
    int n0 = blockIdx.x * 256 + wn * 64;

    // cooperative A-tile staging: thread loads one b128 (8 bf16)
    int arow = tid >> 2;                          // 0..63
    int acol = (tid & 3) * 8;                     // 0,8,16,24
    const unsigned short* Abase = A + (size_t)(m0 + arow) * K + acol;

    v8f acc[2][4];
#pragma unroll
    for (int mt = 0; mt < 2; ++mt)
#pragma unroll
        for (int nt = 0; nt < 4; ++nt) acc[mt][nt] = {};

    v16bf bA[4], bB[4];
    const unsigned short* ldsArd = &ldsA[0][(wm * 32 + cl) * 32 + hi * 8];

    // ---- prologue: stage A(k=0) into ldsA[0], preload B(k=0) into bA
    *(u32x4*)(&ldsA[0][arow * 32 + acol]) = *(const u32x4*)(Abase);
#pragma unroll
    for (int nt = 0; nt < 4; ++nt)
        bA[nt] = load_bfrag(BT, K, n0 + nt * 16 + cl, 0, lane);
    __syncthreads();

    for (int kk = 0; kk < K; kk += 64) {
        // ---- half 1: compute (ldsA[0], bA) @ kk; stage kk+32 -> (ldsA[1], bB)
        *(u32x4*)(&ldsA[1][arow * 32 + acol]) = *(const u32x4*)(Abase + kk + 32);
#pragma unroll
        for (int nt = 0; nt < 4; ++nt)
            bB[nt] = load_bfrag(BT, K, n0 + nt * 16 + cl, kk + 32, lane);
        __builtin_prefetch(BT + (size_t)(n0 + cl) * K + kk + 64, 0, 1);
#pragma unroll
        for (int mt = 0; mt < 2; ++mt) {
            FragU a;
            const unsigned short* p = ldsArd + mt * 16 * 32;
            a.q[0] = *(const u32x4*)(p);
            a.q[1] = *(const u32x4*)(p + 16);
#pragma unroll
            for (int nt = 0; nt < 4; ++nt)
                acc[mt][nt] = __builtin_amdgcn_wmma_f32_16x16x32_bf16(
                    false, a.v, false, bA[nt], (short)0, acc[mt][nt], false, false);
        }
        __syncthreads();

        // ---- half 2: compute (ldsA[1], bB) @ kk+32; stage kk+64 -> (ldsA[0], bA)
        if (kk + 64 < K) {   // block-uniform
            *(u32x4*)(&ldsA[0][arow * 32 + acol]) = *(const u32x4*)(Abase + kk + 64);
#pragma unroll
            for (int nt = 0; nt < 4; ++nt)
                bA[nt] = load_bfrag(BT, K, n0 + nt * 16 + cl, kk + 64, lane);
            __builtin_prefetch(BT + (size_t)(n0 + cl) * K + kk + 96, 0, 1);
        }
#pragma unroll
        for (int mt = 0; mt < 2; ++mt) {
            FragU a;
            const unsigned short* p = ldsArd + (64 * 32) + mt * 16 * 32;  // ldsA[1]
            a.q[0] = *(const u32x4*)(p);
            a.q[1] = *(const u32x4*)(p + 16);
#pragma unroll
            for (int nt = 0; nt < 4; ++nt)
                acc[mt][nt] = __builtin_amdgcn_wmma_f32_16x16x32_bf16(
                    false, a.v, false, bB[nt], (short)0, acc[mt][nt], false, false);
        }
        __syncthreads();
    }

    // ---- epilogue
#pragma unroll
    for (int mt = 0; mt < 2; ++mt)
#pragma unroll
        for (int nt = 0; nt < 4; ++nt) {
            int n = n0 + nt * 16 + cl;
            float bv = bias ? bscale * bias[n] : 0.0f;
#pragma unroll
            for (int j = 0; j < 8; ++j) {
                int row = m0 + wm * 32 + mt * 16 + j + 8 * hi;
                Cout[(size_t)row * N + n] = acc[mt][nt][j] + bv;
            }
        }
}

// ---------------- flash attention over one cache tier ----------------
// Qbf : bf16 [B*S, HID] row-major   (per head: cols h*64..h*64+63)
// Kbf : bf16 [C,   HID] row-major
// VT  : bf16 [HID, C]   row-major   (transposed values)
// biasC : f32 [C]   (-0.1*age + 0.05*access)
// ctx : f32 [B*S, HID];  accum ? += : =
// grid (S/128, B*NH), block 256 (8 waves, one 16-row query tile each)
__global__ __launch_bounds__(256)
void k_flash_wmma(const unsigned short* __restrict__ Qbf,
                  const unsigned short* __restrict__ Kbf,
                  const unsigned short* __restrict__ VT,
                  const float* __restrict__ biasC,
                  float* __restrict__ ctx, int C, int accum) {
    __shared__ unsigned short ldsP[8][16 * 40];   // 16x32 P tile per wave, stride 40

    int wave = threadIdx.x >> 5;
    int lane = threadIdx.x & 31;
    int hi = lane >> 4;
    int cl = lane & 15;
    int bh = blockIdx.y;
    int b = bh / NHH, h = bh - b * NHH;
    int s0 = blockIdx.x * 128 + wave * 16;
    int qrow = b * SS + s0 + cl;                  // A-frag row for this lane
    int hbase = h * HDD;

    float m_run[8], l_run[8];
    v8f o0 = {}, o1 = {}, o2 = {}, o3 = {};
#pragma unroll
    for (int j = 0; j < 8; ++j) { m_run[j] = -1e30f; l_run[j] = 0.0f; }

    unsigned short* lds = &ldsP[wave][0];

    for (int c0 = 0; c0 < C; c0 += 32) {
        // ---- scores: Q[16,64] @ K^T -> two 16x16 f32 tiles (cols c0.., c0+16..)
        v8f sc0 = {}, sc1 = {};
#pragma unroll
        for (int kk = 0; kk < HDD; kk += 32) {
            v16bf a = load_afrag(Qbf, HIDD, qrow, hbase + kk, lane);
            // B-frag: lane col n -> cache row c0+hf*16+n ; K = hbase+kk+8*hi+i (i<8), +16+8*hi+(i-8) (i>=8)
            const unsigned short* p0 = Kbf + (size_t)(c0 + cl) * HIDD + hbase + kk + hi * 8;
            const unsigned short* p1 = Kbf + (size_t)(c0 + 16 + cl) * HIDD + hbase + kk + hi * 8;
            FragU u0, u1;
            u0.q[0] = *(const u32x4*)(p0); u0.q[1] = *(const u32x4*)(p0 + 16);
            u1.q[0] = *(const u32x4*)(p1); u1.q[1] = *(const u32x4*)(p1 + 16);
            sc0 = __builtin_amdgcn_wmma_f32_16x16x32_bf16(false, a, false, u0.v, (short)0, sc0, false, false);
            sc1 = __builtin_amdgcn_wmma_f32_16x16x32_bf16(false, a, false, u1.v, (short)0, sc1, false, false);
        }

        float bias0 = biasC[c0 + cl];
        float bias1 = biasC[c0 + 16 + cl];

        __syncthreads();   // ldsP free from previous chunk across the block

        // ---- online softmax row ops (row = j + 8*hi, spread over a 16-lane half)
        float fac[8];
#pragma unroll
        for (int j = 0; j < 8; ++j) {
            float sA = sc0[j] * 0.125f + bias0;     // 1/sqrt(64)
            float sB = sc1[j] * 0.125f + bias1;
            float v = fmaxf(sA, sB);
            v = fmaxf(v, __shfl_xor(v, 8, 32));
            v = fmaxf(v, __shfl_xor(v, 4, 32));
            v = fmaxf(v, __shfl_xor(v, 2, 32));
            v = fmaxf(v, __shfl_xor(v, 1, 32));
            float mn = fmaxf(m_run[j], v);
            fac[j] = __expf(m_run[j] - mn);
            m_run[j] = mn;
            float p0 = __expf(sA - mn);
            float p1 = __expf(sB - mn);
            int row = j + 8 * hi;
            lds[row * 40 + cl]      = f2bf(p0);
            lds[row * 40 + cl + 16] = f2bf(p1);
            float s = p0 + p1;
            s += __shfl_xor(s, 8, 32);
            s += __shfl_xor(s, 4, 32);
            s += __shfl_xor(s, 2, 32);
            s += __shfl_xor(s, 1, 32);
            l_run[j] = l_run[j] * fac[j] + s;
        }
#pragma unroll
        for (int j = 0; j < 8; ++j) {
            o0[j] *= fac[j]; o1[j] *= fac[j]; o2[j] *= fac[j]; o3[j] *= fac[j];
        }

        __syncthreads();   // P tile visible

        // ---- P (16x32) as A-frag from LDS
        FragU pa;
        {
            const unsigned short* p = &lds[cl * 40 + hi * 8];
            pa.q[0] = *(const u32x4*)(p);
            pa.q[1] = *(const u32x4*)(p + 16);
        }

        // ---- O += P @ V   (V chunk [32, 64] from VT [HID, C], contiguous rows)
#pragma unroll
        for (int nt = 0; nt < 4; ++nt) {
            const unsigned short* p = VT + (size_t)(hbase + nt * 16 + cl) * C + c0 + hi * 8;
            FragU vb;
            vb.q[0] = *(const u32x4*)(p);
            vb.q[1] = *(const u32x4*)(p + 16);
            if      (nt == 0) o0 = __builtin_amdgcn_wmma_f32_16x16x32_bf16(false, pa.v, false, vb.v, (short)0, o0, false, false);
            else if (nt == 1) o1 = __builtin_amdgcn_wmma_f32_16x16x32_bf16(false, pa.v, false, vb.v, (short)0, o1, false, false);
            else if (nt == 2) o2 = __builtin_amdgcn_wmma_f32_16x16x32_bf16(false, pa.v, false, vb.v, (short)0, o2, false, false);
            else              o3 = __builtin_amdgcn_wmma_f32_16x16x32_bf16(false, pa.v, false, vb.v, (short)0, o3, false, false);
        }
    }

    // ---- epilogue: divide by l, write/accumulate ctx[b*S+row][h*64+col]
    v8f os[4] = {o0, o1, o2, o3};
#pragma unroll
    for (int nt = 0; nt < 4; ++nt) {
#pragma unroll
        for (int j = 0; j < 8; ++j) {
            int row = s0 + j + 8 * hi;
            int col = hbase + nt * 16 + cl;
            float val = os[nt][j] / l_run[j];
            float* dst = ctx + (size_t)(b * SS + row) * HIDD + col;
            if (accum) *dst += val; else *dst = val;
        }
    }
}

// ---------------- LayerNorm over HID=1024, one block per row ----------------
__global__ __launch_bounds__(256)
void k_layernorm(const float* __restrict__ x, const float* __restrict__ g,
                 const float* __restrict__ bta, float* __restrict__ out) {
    int row = blockIdx.x;
    const float* xr = x + (size_t)row * HIDD;
    float v[4], s = 0.f, s2 = 0.f;
#pragma unroll
    for (int i = 0; i < 4; ++i) {
        v[i] = xr[threadIdx.x + i * 256];
        s += v[i]; s2 += v[i] * v[i];
    }
#pragma unroll
    for (int off = 16; off; off >>= 1) {
        s  += __shfl_xor(s, off, 32);
        s2 += __shfl_xor(s2, off, 32);
    }
    __shared__ float ss[8], ss2[8];
    if ((threadIdx.x & 31) == 0) { ss[threadIdx.x >> 5] = s; ss2[threadIdx.x >> 5] = s2; }
    __syncthreads();
    float ts = 0.f, ts2 = 0.f;
#pragma unroll
    for (int i = 0; i < 8; ++i) { ts += ss[i]; ts2 += ss2[i]; }
    float mu = ts * (1.0f / HIDD);
    float var = ts2 * (1.0f / HIDD) - mu * mu;
    float inv = rsqrtf(var + 1e-5f);
#pragma unroll
    for (int i = 0; i < 4; ++i) {
        int c = threadIdx.x + i * 256;
        out[(size_t)row * HIDD + c] = (v[i] - mu) * inv * g[c] + bta[c];
    }
}

// ---------------- host orchestration ----------------
static inline int cdiv(int a, int b) { return (a + b - 1) / b; }

extern "C" void kernel_launch(void* const* d_in, const int* in_sizes, int n_in,
                              void* d_out, int out_size, void* d_ws, size_t ws_size,
                              hipStream_t stream) {
    (void)in_sizes; (void)n_in; (void)out_size; (void)ws_size;

    const float* inputs      = (const float*)d_in[0];
    const float* hot_keys    = (const float*)d_in[1];
    const float* hot_values  = (const float*)d_in[2];
    const float* hot_age     = (const float*)d_in[3];
    const float* hot_access  = (const float*)d_in[4];
    const float* cold_keys   = (const float*)d_in[5];
    const float* cold_values = (const float*)d_in[6];
    const float* cold_age    = (const float*)d_in[7];
    const float* cold_access = (const float*)d_in[8];
    const float* Wq = (const float*)d_in[9];  const float* bq = (const float*)d_in[10];
    const float* Wk = (const float*)d_in[11]; const float* bk = (const float*)d_in[12];
    const float* Wv = (const float*)d_in[13]; const float* bv = (const float*)d_in[14];
    const float* Wo = (const float*)d_in[15]; const float* bo = (const float*)d_in[16];
    const float* Wc = (const float*)d_in[17]; const float* bc = (const float*)d_in[18];
    const float* Wd = (const float*)d_in[19]; const float* bd = (const float*)d_in[20];
    const float* gamma = (const float*)d_in[21];
    const float* beta  = (const float*)d_in[22];

    // ---- workspace carve ----
    char* p = (char*)d_ws;
    auto alloc = [&](size_t bytes) { char* r = p; p += (bytes + 255) & ~(size_t)255; return r; };
    const int ROWS = BB * SS;                               // 2048

    unsigned short* x_bf    = (unsigned short*)alloc((size_t)ROWS * HIDD * 2);
    unsigned short* WqT     = (unsigned short*)alloc((size_t)HIDD * HIDD * 2);
    unsigned short* WkT     = (unsigned short*)alloc((size_t)HIDD * HIDD * 2);
    unsigned short* WvT     = (unsigned short*)alloc((size_t)HIDD * HIDD * 2);
    unsigned short* WoT     = (unsigned short*)alloc((size_t)HIDD * HIDD * 2);
    unsigned short* WcT     = (unsigned short*)alloc((size_t)COMPN * HIDD * 2);
    unsigned short* WdT     = (unsigned short*)alloc((size_t)HIDD * COMPN * 2);
    unsigned short* hotk_bf = (unsigned short*)alloc((size_t)HOTN * HIDD * 2);
    unsigned short* hotv_bf = (unsigned short*)alloc((size_t)HOTN * HIDD * 2);
    unsigned short* coldk_bf= (unsigned short*)alloc((size_t)COLDN * HIDD * 2);
    unsigned short* coldv_bf= (unsigned short*)alloc((size_t)COLDN * HIDD * 2);
    unsigned short* Q_bf    = (unsigned short*)alloc((size_t)ROWS * HIDD * 2);
    unsigned short* Khot_bf = (unsigned short*)alloc((size_t)HOTN * HIDD * 2);
    unsigned short* Kcold_bf= (unsigned short*)alloc((size_t)COLDN * HIDD * 2);
    unsigned short* VhotT   = (unsigned short*)alloc((size_t)HIDD * HOTN * 2);
    unsigned short* VcoldT  = (unsigned short*)alloc((size_t)HIDD * COLDN * 2);
    unsigned short* Ccold_bf= (unsigned short*)alloc((size_t)COLDN * COMPN * 2);
    unsigned short* ctx_bf  = (unsigned short*)alloc((size_t)ROWS * HIDD * 2);
    float* ctxf   = (float*)alloc((size_t)ROWS * HIDD * 4);
    float* tmpf   = (float*)alloc((size_t)COLDN * HIDD * 4);   // reused f32 GEMM staging
    float* bias_h = (float*)alloc((size_t)HOTN * 4);
    float* bias_c = (float*)alloc((size_t)COLDN * 4);

    auto cvt = [&](const float* src, unsigned short* dst, int n) {
        k_f32_to_bf16<<<cdiv(n, 256), 256, 0, stream>>>(src, dst, n);
    };
    auto tr = [&](const float* src, unsigned short* dst, int R, int C) {
        k_transpose_bf16<<<cdiv(R * C, 256), 256, 0, stream>>>(src, dst, R, C);
    };
    auto gemm = [&](const unsigned short* A, const unsigned short* BT, const float* bias,
                    float bscale, float* Cf, int M, int N, int K) {
        dim3 g(N / 256, M / 64);
        k_gemm_bf16<<<g, 256, 0, stream>>>(A, BT, bias, bscale, Cf, M, N, K);
    };

    // 0) per-tier score biases
    k_bias_combine<<<cdiv(HOTN, 256), 256, 0, stream>>>(hot_age, hot_access, bias_h, HOTN);
    k_bias_combine<<<cdiv(COLDN, 256), 256, 0, stream>>>(cold_age, cold_access, bias_c, COLDN);

    // 1) activation converts + weight transposes
    cvt(inputs, x_bf, ROWS * HIDD);
    cvt(hot_keys, hotk_bf, HOTN * HIDD);
    cvt(hot_values, hotv_bf, HOTN * HIDD);
    cvt(cold_keys, coldk_bf, COLDN * HIDD);
    cvt(cold_values, coldv_bf, COLDN * HIDD);
    tr(Wq, WqT, HIDD, HIDD);
    tr(Wk, WkT, HIDD, HIDD);
    tr(Wv, WvT, HIDD, HIDD);
    tr(Wo, WoT, HIDD, HIDD);
    tr(Wc, WcT, HIDD, COMPN);     // Wc [1024,512] -> WcT [512,1024]
    tr(Wd, WdT, COMPN, HIDD);     // Wd [512,1024] -> WdT [1024,512]

    // 2) projections (bf16 WMMA GEMMs, f32 staging then convert)
    gemm(x_bf, WqT, bq, 1.0f, tmpf, ROWS, HIDD, HIDD);
    cvt(tmpf, Q_bf, ROWS * HIDD);

    gemm(hotk_bf, WkT, bk, 1.0f, tmpf, HOTN, HIDD, HIDD);
    cvt(tmpf, Khot_bf, HOTN * HIDD);

    gemm(coldk_bf, WkT, bk, 1.0f, tmpf, COLDN, HIDD, HIDD);
    cvt(tmpf, Kcold_bf, COLDN * HIDD);

    gemm(hotv_bf, WvT, bv, 1.0f, tmpf, HOTN, HIDD, HIDD);
    tr(tmpf, VhotT, HOTN, HIDD);                           // -> [HID, HOT]

    gemm(coldv_bf, WcT, bc, 1.0f, tmpf, COLDN, COMPN, HIDD);
    cvt(tmpf, Ccold_bf, COLDN * COMPN);
    gemm(Ccold_bf, WdT, bd, 1.0f, tmpf, COLDN, HIDD, COMPN);
    tr(tmpf, VcoldT, COLDN, HIDD);                         // -> [HID, COLD]

    // 3) flash attention per tier (hot writes, cold accumulates)
    {
        dim3 g(SS / 128, BB * NHH);
        k_flash_wmma<<<g, 256, 0, stream>>>(Q_bf, Khot_bf, VhotT, bias_h, ctxf, HOTN, 0);
        k_flash_wmma<<<g, 256, 0, stream>>>(Q_bf, Kcold_bf, VcoldT, bias_c, ctxf, COLDN, 1);
    }

    // 4) output projection: (hot+cold) @ Wo + 2*bo  (linearity folds both bo adds)
    cvt(ctxf, ctx_bf, ROWS * HIDD);
    gemm(ctx_bf, WoT, bo, 2.0f, tmpf, ROWS, HIDD, HIDD);

    // 5) LayerNorm -> d_out
    k_layernorm<<<ROWS, 256, 0, stream>>>(tmpf, gamma, beta, (float*)d_out);
}
